// TAModule_39848706573414
// MI455X (gfx1250) — matrix operand order repaired and checked by reading, not compiled, not run on hardware
//
#include <hip/hip_runtime.h>
#include <hip/hip_bf16.h>
#include <math.h>

typedef __attribute__((ext_vector_type(16))) _Float16 v16h;
typedef __attribute__((ext_vector_type(8)))  float    v8f;
typedef __attribute__((ext_vector_type(4)))  int      v4i;

#define AS1 __attribute__((address_space(1)))
#define AS3 __attribute__((address_space(3)))

#if __has_builtin(__builtin_amdgcn_global_load_async_to_lds_b128) && \
    __has_builtin(__builtin_amdgcn_s_wait_asynccnt)
#define HAS_ASYNC_LDS 1
#else
#define HAS_ASYNC_LDS 0
#endif

#define BB 8
#define LL 8
#define CC 64
#define HH 56
#define WW 56
#define HWp (HH*WW)          // 3136
#define NIMG (BB*LL)         // 64
#define KTOT (CC*9)          // 576
#define KC 32
#define NKC (KTOT/KC)        // 18
#define TPIX 128
#define NTILES ((HWp + TPIX - 1)/TPIX)  // 25

// workspace float offsets
#define WS_POOLED 0          // [B][C][L]     4096
#define WS_SGAIN  4096       // [n][ci]       4096
#define WS_FBIAS  8192       // [n][co]       4096
#define WS_PSUM   12288      // [C][8]        512
#define WS_PSUMSQ 12800      // [C][8]        512
#define WS_SCALE  13312      // [C]           64
#define WS_SHIFT  13376      // [C]           64

// fragment indexing: [tile(4)][kc(2)][lane(32)][e(16)]
__device__ __forceinline__ int fidx(int tile, int kc, int lane, int e) {
  return ((tile*2 + kc)*32 + lane)*16 + e;
}

// ---------------------------------------------------------------------------
// Kernel 1: Fourier attention + pooling.  One WG per (b,c).
// Spatial 56x56 DFT (padded to 64) done as f16 WMMA GEMMs w/ f32 accum:
//   stage1: S1 = W   * X      (W complex const, X real frame)
//   stage2: F  = S1  * W^T    (complex*complex as 4 real GEMMs)
// All 8 frames' spectra resident in LDS (271 KB; CDNA5 WGP has 320 KB).
// Frame staging uses GLOBAL_LOAD_ASYNC_TO_LDS_B128 (ASYNCcnt) when available.
// Then 8-pt temporal DFT + |.| + fused pooling (VALU).
// pooled[b,c,l] = mean_hw( x * (1 + 0.01*|FFT3(x)|) )   (ortho norms)
// ---------------------------------------------------------------------------
__global__ __launch_bounds__(256) void fourier_pool_kernel(
    const float* __restrict__ x, float* __restrict__ pooled)
{
  extern __shared__ __align__(128) char smem_raw[];
  _Float16* WrA  = (_Float16*)smem_raw;     // [4][2][32][16] A-layout of Wr
  _Float16* WiA  = WrA  + 4096;             // A-layout of Wi
  _Float16* WrB  = WiA  + 4096;             // B-layout of W^T (re)
  _Float16* WiB  = WrB  + 4096;             // B-layout of W^T (im)
  _Float16* Xb   = WiB  + 4096;             // B-layout of current frame
  _Float16* S1rA = Xb   + 4096;             // A-layout of stage-1 re
  _Float16* S1iA = S1rA + 4096;             // A-layout of stage-1 im
  float* SpR = (float*)(S1iA + 4096);       // [8][3136] spectra re
  float* SpI = SpR + LL*HWp;                // [8][3136] spectra im
  float* Fx  = SpI + LL*HWp;                // [3136] frame staging / reduce
  float* Twr = Fx  + HWp;                   // [64] temporal twiddles re
  float* Twi = Twr + 64;                    // [64]

  const int tid  = threadIdx.x;
  const int lane = tid & 31;
  const int wv   = tid >> 5;
  const int bc   = blockIdx.x;
  const int b    = bc / CC;
  const int c    = bc % CC;

  const float inv56 = 0.133630620956212f;   // 1/sqrt(56)
  // Build DFT-matrix fragments (once).  A-layout (16-bit A 16x32, ISA 7.12.2):
  // lanes 0-15 hold K 0-7,16-23; lanes 16-31 hold K 8-15,24-31.
  // B-layout: same K order as A (lanes 0-15 K 0-7,16-23; lanes 16-31 K 8-15,24-31).
  for (int fi = tid; fi < 4096; fi += 256) {
    int e  = fi & 15;
    int ln = (fi >> 4) & 31;
    int kc = (fi >> 9) & 1;
    int t  = fi >> 10;
    // ---- A-layout entry: row m=u, K=h
    int m  = t*16 + (ln & 15);
    int klA = (e & 7) + ((e & 8) ? 16 : 0) + ((ln & 16) ? 8 : 0);
    int h  = kc*KC + klA;
    float wr = 0.f, wi = 0.f;
    if (m < HH && h < WW) {
      int md = (m * h) % 56;
      float ang = -6.283185307179586f * (float)md * (1.0f/56.0f);
      wr = cosf(ang) * inv56; wi = sinf(ang) * inv56;
    }
    WrA[fi] = (_Float16)wr; WiA[fi] = (_Float16)wi;
    // ---- B-layout entry: column n=v, K=w  (B[w][v] = W[v][w])
    int v  = t*16 + (ln & 15);
    int klB = (e & 7) + ((e & 8) ? 16 : 0) + ((ln & 16) ? 8 : 0);
    int w  = kc*KC + klB;
    float br = 0.f, bi = 0.f;
    if (v < HH && w < WW) {
      int md = (v * w) % 56;
      float ang = -6.283185307179586f * (float)md * (1.0f/56.0f);
      br = cosf(ang) * inv56; bi = sinf(ang) * inv56;
    }
    WrB[fi] = (_Float16)br; WiB[fi] = (_Float16)bi;
  }
  const float inv8 = 0.3535533905932738f;   // 1/sqrt(8)
  if (tid < 64) {
    int lu = tid / 8, l = tid % 8;
    int md = (lu * l) & 7;
    float ang = -6.283185307179586f * (float)md * 0.125f;
    Twr[tid] = cosf(ang) * inv8;
    Twi[tid] = sinf(ang) * inv8;
  }
  __syncthreads();

  for (int l = 0; l < LL; ++l) {
    const float* frame = x + (size_t)((b*LL + l)*CC + c) * HWp;
#if HAS_ASYNC_LDS
    // CDNA5 async DMA: global -> LDS directly, tracked by ASYNCcnt.
    // 3136 floats = 784 x b128; 16B-aligned on both sides.
    for (int i4 = tid; i4 < HWp/4; i4 += 256)
      __builtin_amdgcn_global_load_async_to_lds_b128(
          (AS1 v4i*)(frame + 4*i4), (AS3 v4i*)(Fx + 4*i4), 0, 0);
    __builtin_amdgcn_s_wait_asynccnt(0);
#else
    for (int i = tid; i < HWp; i += 256) Fx[i] = frame[i];   // coalesced
#endif
    __syncthreads();
    // frame -> B-fragment layout (pad 56->64 with zeros)
    for (int fi = tid; fi < 4096; fi += 256) {
      int e  = fi & 15;
      int ln = (fi >> 4) & 31;
      int kc = (fi >> 9) & 1;
      int t  = fi >> 10;
      int w  = t*16 + (ln & 15);
      int h  = kc*KC + (e & 7) + ((e & 8) ? 16 : 0) + ((ln & 16) ? 8 : 0);
      float v = (h < HH && w < WW) ? Fx[h*WW + w] : 0.f;
      Xb[fi] = (_Float16)v;
    }
    __syncthreads();
    // stage 1: S1 = W * X ; 32 jobs = 16 (mt,nt) tiles x {re,im}
    for (int jj = 0; jj < 4; ++jj) {
      int j = wv + jj*8;
      int which = j >> 4;            // 0: re (Wr), 1: im (Wi)
      int tile  = j & 15;
      int mt = tile >> 2, nt = tile & 3;
      const _Float16* Asrc = which ? WiA : WrA;
      v8f acc = {};
      for (int kc = 0; kc < 2; ++kc) {
        v16h a = *(const v16h*)&Asrc[fidx(mt, kc, lane, 0)];
        v16h bm = *(const v16h*)&Xb[fidx(nt, kc, lane, 0)];
        acc = __builtin_amdgcn_wmma_f32_16x16x32_f16(
            false, a, false, bm, (short)0, acc, false, false);
      }
      // scatter D (C/D layout) into A-fragment layout for stage 2
      _Float16* Ds = which ? S1iA : S1rA;
      for (int r = 0; r < 8; ++r) {
        int m  = mt*16 + r + ((lane & 16) ? 8 : 0);
        int k2 = nt*16 + (lane & 15);
        int kc2 = k2 >> 5, kl2 = k2 & 31;
        int ln2 = (m & 15) + ((kl2 & 8) ? 16 : 0);
        int e2  = (kl2 & 7) + ((kl2 & 16) ? 8 : 0);
        Ds[fidx(m >> 4, kc2, ln2, e2)] = (_Float16)acc[r];
      }
    }
    __syncthreads();
    // stage 2: F = S1 * W^T (complex):
    //   re = S1r*WrT - S1i*WiT ;  im = S1r*WiT + S1i*WrT
    for (int jj = 0; jj < 4; ++jj) {
      int j = wv + jj*8;
      int which = j >> 4;            // 0: re, 1: im
      int tile  = j & 15;
      int mt = tile >> 2, nt = tile & 3;
      v8f acc1 = {}, acc2 = {};
      for (int kc = 0; kc < 2; ++kc) {
        v16h a1 = *(const v16h*)&S1rA[fidx(mt, kc, lane, 0)];
        v16h b1 = *(const v16h*)&(which ? WiB : WrB)[fidx(nt, kc, lane, 0)];
        acc1 = __builtin_amdgcn_wmma_f32_16x16x32_f16(
            false, a1, false, b1, (short)0, acc1, false, false);
      }
      for (int kc = 0; kc < 2; ++kc) {
        v16h a2 = *(const v16h*)&S1iA[fidx(mt, kc, lane, 0)];
        v16h b2 = *(const v16h*)&(which ? WrB : WiB)[fidx(nt, kc, lane, 0)];
        acc2 = __builtin_amdgcn_wmma_f32_16x16x32_f16(
            false, a2, false, b2, (short)0, acc2, false, false);
      }
      v8f res = which ? (acc1 + acc2) : (acc1 - acc2);
      float* Sp = which ? SpI : SpR;
      for (int r = 0; r < 8; ++r) {
        int m = mt*16 + r + ((lane & 16) ? 8 : 0);
        int v = nt*16 + (lane & 15);
        if (m < HH && v < WW) Sp[l*HWp + m*WW + v] = res[r];
      }
    }
    __syncthreads();
  }

  // temporal 8-pt DFT per pixel, magnitude, fused pooling
  float acc[LL];
  for (int l = 0; l < LL; ++l) acc[l] = 0.f;
  for (int i = tid; i < HWp; i += 256) {
    float sr[LL], si[LL], xv[LL];
    for (int l = 0; l < LL; ++l) {
      sr[l] = SpR[l*HWp + i];
      si[l] = SpI[l*HWp + i];
      xv[l] = x[(size_t)((b*LL + l)*CC + c) * HWp + i];
    }
    for (int lu = 0; lu < LL; ++lu) {
      float fr = 0.f, fi2 = 0.f;
      for (int l = 0; l < LL; ++l) {
        float tr = Twr[lu*8 + l], ti = Twi[lu*8 + l];
        fr  = fmaf(tr, sr[l], fr);  fr  = fmaf(-ti, si[l], fr);
        fi2 = fmaf(tr, si[l], fi2); fi2 = fmaf( ti, sr[l], fi2);
      }
      float mag = sqrtf(fr*fr + fi2*fi2);
      acc[lu] = fmaf(xv[lu], 1.0f + 0.01f*mag, acc[lu]);
    }
  }
  __syncthreads();
  for (int l = 0; l < LL; ++l) Fx[l*256 + tid] = acc[l];   // reuse staging
  __syncthreads();
  if (tid < LL) {
    float s = 0.f;
    for (int t = 0; t < 256; ++t) s += Fx[tid*256 + t];    // fixed order
    pooled[(b*CC + c)*LL + tid] = s * (1.0f/(float)HWp);
  }
}

// ---------------------------------------------------------------------------
// Kernel 2: calibration GEMMs -> per-(n,ci) gain and per-(n,co) bias
// ---------------------------------------------------------------------------
__global__ __launch_bounds__(256) void calib_fc_kernel(
    const float* __restrict__ pooled,
    const float* __restrict__ tw, const float* __restrict__ tb,
    const float* __restrict__ fcw, const float* __restrict__ fcb,
    const float* __restrict__ convb,
    float* __restrict__ sgain, float* __restrict__ fbias)
{
  __shared__ float fco[NIMG];
  const int tid = threadIdx.x;
  for (int i = tid; i < NIMG*CC; i += 256) {
    int n = i / CC, o = i % CC;
    int b = n / LL, l = n % LL;
    float s = tb[o];
    for (int c2 = 0; c2 < CC; ++c2)
      s = fmaf(tw[o*CC + c2], pooled[(b*CC + c2)*LL + l], s);
    sgain[i] = s + 1.0f;                 // (calib + 1), indexed by ci
  }
  if (tid < NIMG) {
    int b = tid / LL, l = tid % LL;
    float s = fcb[0];
    for (int c2 = 0; c2 < CC; ++c2)
      s = fmaf(fcw[c2], pooled[(b*CC + c2)*LL + l], s);
    fco[tid] = s + 1.0f;                 // (fcout + 1)
  }
  __syncthreads();
  for (int i = tid; i < NIMG*CC; i += 256) {
    int n = i / CC, o = i % CC;
    fbias[i] = convb[o] * fco[n];
  }
}

// ---------------------------------------------------------------------------
// Kernel 3: dynamic conv as implicit GEMM with v_wmma_f32_16x16x32_f16.
// fw[n,co,ci,kh,kw] = conv_w * s[n,ci]  ==>  conv(conv_w, s[n,ci]*x[n,ci]).
// Per WG: one image, 128-pixel tile. A = weights 64x576 (LDS, A-frag layout),
// B = scaled im2col patches, built per 32-K chunk, double-buffered.
// Writes y = x + conv + bias to d_out (pre-BN buffer).
// ---------------------------------------------------------------------------
__global__ __launch_bounds__(256) void dynconv_wmma_kernel(
    const float* __restrict__ x, const float* __restrict__ convw,
    const float* __restrict__ sgain, const float* __restrict__ fbias,
    float* __restrict__ y)
{
  extern __shared__ __align__(128) char smem_raw[];
  _Float16* Afrag = (_Float16*)smem_raw;            // [4][18][32][16]  73728 B
  _Float16* Bfrag = Afrag + 4*NKC*32*16;            // [2][8][32][16]   16384 B
  float* sg = (float*)(Bfrag + 2*8*32*16);          // [64]
  float* fb = sg + CC;                              // [64]

  const int tid  = threadIdx.x;
  const int lane = tid & 31;
  const int wv   = tid >> 5;            // 8 waves; wave = 16-pixel subtile
  const int n    = blockIdx.x;
  const int pixbase = blockIdx.y * TPIX;

  if (tid < CC) { sg[tid] = sgain[n*CC + tid]; fb[tid] = fbias[n*CC + tid]; }

  // A fragments: per-lane WMMA A layout (ISA 7.12.2, 16-bit A 16x32)
  for (int fi = tid; fi < 4*NKC*32*16; fi += 256) {
    int e  = fi & 15;
    int ln = (fi >> 4) & 31;
    int kc = (fi >> 9) % NKC;
    int mt = fi / (512*NKC);
    int m  = mt*16 + (ln & 15);
    int kl = (e & 7) + ((e & 8) ? 16 : 0) + ((ln & 16) ? 8 : 0);
    Afrag[fi] = (_Float16)convw[m*KTOT + kc*KC + kl];
  }
  __syncthreads();

  const float* xin = x + (size_t)n * CC * HWp;

  v8f acc[4] = {};

  // Build one 32(K) x 128(pix) B chunk directly into B-frag layout.
  auto buildB = [&](int buf, int kc) {
    _Float16* Bb = Bfrag + buf * (8*32*16);
    for (int fi = tid; fi < KC*TPIX; fi += 256) {
      int pix = fi & (TPIX-1);
      int kl  = fi >> 7;
      int k   = kc*KC + kl;
      int ci  = k / 9;
      int r   = k - ci*9;
      int kh  = r / 3;
      int kw  = r - kh*3;
      int pg  = pixbase + pix;
      float v = 0.f;
      if (pg < HWp) {
        int h = pg / WW, wc = pg - h*WW;
        int hh = h + kh - 1, wx = wc + kw - 1;
        if (hh >= 0 && hh < HH && wx >= 0 && wx < WW)
          v = xin[ci*HWp + hh*WW + wx] * sg[ci];
      }
      int s  = pix >> 4;
      int ln = (pix & 15) + ((kl & 8) ? 16 : 0);
      int e  = (kl & 7) + ((kl & 16) ? 8 : 0);
      Bb[(s*32 + ln)*16 + e] = (_Float16)v;
    }
  };

  buildB(0, 0);
  __syncthreads();

  for (int kc = 0; kc < NKC; ++kc) {
    // prefetch next-next chunk's input rows (global_prefetch_b8)
    if (kc + 2 < NKC && tid < 4) {
      int cpf = ((kc + 2) * KC) / 9 + tid;
      if (cpf < CC) __builtin_prefetch(&xin[cpf*HWp + pixbase], 0, 0);
    }
    if (kc + 1 < NKC) buildB((kc + 1) & 1, kc + 1);
    const int cur = kc & 1;
    v16h bvec = *(const v16h*)&Bfrag[((cur*8 + wv)*32 + lane)*16];
    for (int mt = 0; mt < 4; ++mt) {
      v16h avec = *(const v16h*)&Afrag[((mt*NKC + kc)*32 + lane)*16];
      acc[mt] = __builtin_amdgcn_wmma_f32_16x16x32_f16(
          false, avec, false, bvec, (short)0, acc[mt], false, false);
    }
    __syncthreads();
  }

  // Epilogue: C/D layout = VGPR r: lanes 0-15 M=r, lanes 16-31 M=r+8, N=lane%16
  for (int mt = 0; mt < 4; ++mt) {
    for (int r = 0; r < 8; ++r) {
      int co = mt*16 + r + ((lane & 16) ? 8 : 0);
      int pg = pixbase + wv*16 + (lane & 15);
      if (pg < HWp) {
        size_t off = (size_t)(n*CC + co) * HWp + pg;
        y[off] = acc[mt][r] + fb[co] + x[off];   // +bias +residual
      }
    }
  }
}

// ---------------------------------------------------------------------------
// Kernels 4-6: BatchNorm (deterministic two-level reduce) + SiLU, in place.
// ---------------------------------------------------------------------------
__global__ __launch_bounds__(256) void bn_stats_kernel(
    const float* __restrict__ y, float* __restrict__ psum, float* __restrict__ psumsq)
{
  __shared__ float rs[256], rq[256];
  const int c = blockIdx.x >> 3;
  const int slice = blockIdx.x & 7;
  const int tid = threadIdx.x;
  float s = 0.f, q = 0.f;
  for (int n = slice*8; n < slice*8 + 8; ++n) {
    const float* p = y + (size_t)(n*CC + c) * HWp;
    for (int i = tid; i < HWp; i += 256) {
      float v = p[i];
      s += v; q = fmaf(v, v, q);
    }
  }
  rs[tid] = s; rq[tid] = q;
  __syncthreads();
  for (int st = 128; st > 0; st >>= 1) {
    if (tid < st) { rs[tid] += rs[tid+st]; rq[tid] += rq[tid+st]; }
    __syncthreads();
  }
  if (tid == 0) { psum[c*8 + slice] = rs[0]; psumsq[c*8 + slice] = rq[0]; }
}

__global__ void bn_finalize_kernel(
    const float* __restrict__ psum, const float* __restrict__ psumsq,
    const float* __restrict__ gamma, const float* __restrict__ beta,
    float* __restrict__ scale, float* __restrict__ shift)
{
  int c = threadIdx.x;
  if (c < CC) {
    float s = 0.f, q = 0.f;
    for (int i = 0; i < 8; ++i) { s += psum[c*8 + i]; q += psumsq[c*8 + i]; }
    const float invN = 1.0f / (float)((size_t)NIMG * HWp);
    float mean = s * invN;
    float var  = q * invN - mean*mean;
    float inv  = rsqrtf(var + 1e-5f);
    float sc   = gamma[c] * inv;
    scale[c] = sc;
    shift[c] = beta[c] - mean * sc;
  }
}

__global__ __launch_bounds__(256) void bn_apply_kernel(
    float* __restrict__ y, const float* __restrict__ scale, const float* __restrict__ shift)
{
  size_t idx = (size_t)blockIdx.x * 256 + threadIdx.x;
  int c = (int)((idx / HWp) % CC);
  float v = y[idx] * scale[c] + shift[c];
  y[idx] = v / (1.0f + __expf(-v));      // SiLU
}

// ---------------------------------------------------------------------------
extern "C" void kernel_launch(void* const* d_in, const int* in_sizes, int n_in,
                              void* d_out, int out_size, void* d_ws, size_t ws_size,
                              hipStream_t stream) {
  (void)in_sizes; (void)n_in; (void)out_size; (void)ws_size;
  const float* x     = (const float*)d_in[0];
  const float* tw    = (const float*)d_in[1];
  const float* tb    = (const float*)d_in[2];
  const float* fcw   = (const float*)d_in[3];
  const float* fcb   = (const float*)d_in[4];
  const float* convw = (const float*)d_in[5];
  const float* convb = (const float*)d_in[6];
  const float* gamma = (const float*)d_in[7];
  const float* beta  = (const float*)d_in[8];
  float* out = (float*)d_out;            // doubles as pre-BN y buffer
  float* ws  = (float*)d_ws;

  float* pooled = ws + WS_POOLED;
  float* sgain  = ws + WS_SGAIN;
  float* fbias  = ws + WS_FBIAS;
  float* psum   = ws + WS_PSUM;
  float* psumsq = ws + WS_PSUMSQ;
  float* scale  = ws + WS_SCALE;
  float* shift  = ws + WS_SHIFT;

  // 7 f16 fragment arrays (4096 each) + spectra/staging/twiddles in f32
  // = 57,344 + 213,760 = 271,104 B  (< 320 KB CDNA5 WGP LDS)
  const size_t smem1 = (size_t)7*4096*sizeof(_Float16)
                     + (size_t)(2*LL*HWp + HWp + 128)*sizeof(float);
  fourier_pool_kernel<<<dim3(BB*CC), dim3(256), smem1, stream>>>(x, pooled);

  calib_fc_kernel<<<dim3(1), dim3(256), 0, stream>>>(
      pooled, tw, tb, fcw, fcb, convb, sgain, fbias);

  const size_t smem2 = (size_t)(4*NKC*32*16 + 2*8*32*16) * sizeof(_Float16)
                     + (size_t)2*CC*sizeof(float);      // 90,624 B
  dynconv_wmma_kernel<<<dim3(NIMG, NTILES), dim3(256), smem2, stream>>>(
      x, convw, sgain, fbias, out);

  bn_stats_kernel<<<dim3(CC*8), 256, 0, stream>>>(out, psum, psumsq);
  bn_finalize_kernel<<<dim3(1), dim3(64), 0, stream>>>(
      psum, psumsq, gamma, beta, scale, shift);
  bn_apply_kernel<<<dim3((NIMG*CC*HWp)/256), dim3(256), 0, stream>>>(out, scale, shift);
}
